// DCTblock_21517786153312
// MI455X (gfx1250) — matrix operand order repaired and checked by reading, not compiled, not run on hardware
//
#include <hip/hip_runtime.h>

// ---------------------------------------------------------------------------
// DCT-GCN (LTD-style) forward for MI455X / gfx1250, wave32 + WMMA f16.
// B=512, T=125, NODE=66, DCT=20, HID=256, STAGES=6, T_HIS=25.
// Fused GC kernel: one workgroup per batch element. Activations are kept in
// global memory both as f32 (for BN stats / residuals) and as zero-padded f16
// (produced by the DCT / BN kernels). The GC kernel streams the f16 A-slab
// straight into LDS with GLOBAL_LOAD_ASYNC_TO_LDS_B128 (ASYNCcnt), runs X@W
// on v_wmma_f32_16x16x32_f16, keeps the 66x256 intermediate transposed in
// LDS, then runs att@tmp on WMMA again. All register arrays are constant-
// indexed (templated tile counts) and wave-uniform guards are SGPR-based so
// WMMA executes with full EXEC and no spills.
// ---------------------------------------------------------------------------

typedef __attribute__((ext_vector_type(16))) _Float16 v16h;
typedef __attribute__((ext_vector_type(8)))  _Float16 v8h;
typedef __attribute__((ext_vector_type(8)))  float    v8f;

#define BATCH  512
#define TLEN   125
#define NNODE  66
#define DCTN   20
#define HID    256
#define NSTAGE 6
#define THIS   25
#define CCH    (NNODE * HID)     // 16896 BN channels
#define DPAD   32                // encoder K padded (f16 xd tile width)

#define MPADR  80                // node rows padded (5 WMMA M tiles)
#define MPADK  96                // node dim padded as K for GEMM2 (3 k-steps)
#define LDA    40                // A-slab row stride (halfs), 16B aligned
#define LDT    104               // tmp_T row stride (halfs), 16B aligned

// Load a 16-element f16 fragment as two contiguous 8-half (16B) loads.
__device__ __forceinline__ v16h ld_frag(const _Float16* lo, const _Float16* hi) {
    union { v16h v; v8h h[2]; } u;
    u.h[0] = *(const v8h*)lo;
    u.h[1] = *(const v8h*)hi;
    return u.v;
}

__device__ __forceinline__ v8f wmma_f16(v16h a, v16h b, v8f c) {
    // D = A(16x32 f16) * B(32x16 f16) + C(16x16 f32)
    return __builtin_amdgcn_wmma_f32_16x16x32_f16(
        /*neg_a=*/false, a, /*neg_b=*/false, b,
        /*c_mod=*/(short)0, c, /*reuse_a=*/false, /*reuse_b=*/false);
}

// Async 16B copy global -> LDS (ASYNCcnt-tracked).
__device__ __forceinline__ void async_cp16(unsigned lds_off, const _Float16* gp) {
    asm volatile("global_load_async_to_lds_b128 %0, %1, off"
                 :: "v"(lds_off), "v"(gp) : "memory");
}
__device__ __forceinline__ void async_wait0() {
    asm volatile("s_wait_asynccnt 0x0" ::: "memory");
}

// ---------------------------------------------------------------------------
// Fused GraphConvolution:
//   out[b,n,f] = sum_m att[n,m] * (sum_k in[b,m,k]*W[k,f]) + bias[f]
// in16 : [B, NNODE, Kpad] f16, already zero-padded in k.
// NTILES = Nout/16 (16 for HID outputs, 2 for decoder). grid.x=B, 256 thr.
// ---------------------------------------------------------------------------
template <int NTILES>
__global__ __launch_bounds__(256)
void gc_kernel(const _Float16* __restrict__ in16, int Kpad,
               const _Float16* __restrict__ wT,     // [NTILES*16][Kpad] f16
               const _Float16* __restrict__ attP,   // [80][96] f16 zero-padded
               const float* __restrict__ bias,
               float* __restrict__ out, int ldout, int Nstore)
{
    constexpr int NOUT = NTILES * 16;
    constexpr int JMAX = (NTILES + 7) / 8;           // N tiles per wave (2 or 1)

    __shared__ __align__(16) _Float16 Ah[MPADR * LDA];   //  6.4 KB A k-slab
    __shared__ __align__(16) _Float16 Tt[NOUT * LDT];    // tmp^T [f][m] f16

    const int tid  = threadIdx.x;
    const int lane = tid & 31;
    // wave id as an SGPR -> tile guards are scalar branches, EXEC stays full.
    const int w    = __builtin_amdgcn_readfirstlane(threadIdx.x >> 5) & 7;
    const int l15  = lane & 15;
    const int g    = lane >> 4;
    const int b    = blockIdx.x;

    v8h hz;
#pragma unroll
    for (int r = 0; r < 8; ++r) hz[r] = (_Float16)0.f;

    // zero A-slab padding rows 66..79 once (never overwritten by async copies)
    if (tid < 14 * 5) {
        int r = 66 + tid / 5, q = tid % 5;
        *(v8h*)&Ah[r * LDA + q * 8] = hz;
    }
    // zero tmp_T K-padding columns m=80..95 (GEMM2 reads them)
    for (int i = tid; i < NOUT * 2; i += 256) {
        int row = i >> 1, q = i & 1;
        *(v8h*)&Tt[row * LDT + 80 + q * 8] = hz;
    }

    v8f acc[5][JMAX];
#pragma unroll
    for (int mt = 0; mt < 5; ++mt)
#pragma unroll
        for (int j = 0; j < JMAX; ++j)
#pragma unroll
            for (int r = 0; r < 8; ++r) acc[mt][j][r] = 0.f;

    // ---------------- GEMM1: tmp[m][f] = in[b] @ W ----------------
    const _Float16* inb = in16 + (size_t)b * NNODE * Kpad;
    const unsigned  ahBase = (unsigned)(unsigned long long)(void*)&Ah[0];
    const int nks1 = Kpad >> 5;
    for (int ks = 0; ks < nks1; ++ks) {
        __syncthreads();                 // previous k-step done reading Ah
        const int kb = ks << 5;
        // async-copy 66 rows x 32 halfs (64B/row) into the slab: 264 x 16B
        {
            int r = tid >> 2, q = tid & 3;
            async_cp16(ahBase + (unsigned)(r * LDA + q * 8) * 2u,
                       inb + (size_t)r * Kpad + kb + q * 8);
            if (tid < 264 - 256) {
                int s2 = tid + 256;
                int r2 = s2 >> 2, q2 = s2 & 3;
                async_cp16(ahBase + (unsigned)(r2 * LDA + q2 * 8) * 2u,
                           inb + (size_t)r2 * Kpad + kb + q2 * 8);
            }
        }
        async_wait0();
        __syncthreads();

        v16h bf[JMAX];
#pragma unroll
        for (int j = 0; j < JMAX; ++j) {
            if (w + 8 * j < NTILES) {
                const int nt = w + 8 * j;
                // B frag: col nt*16+l15 of W = row of wT, k g*8..+8 and 16+g*8..+8
                const _Float16* bp = wT + (size_t)(nt * 16 + l15) * Kpad + kb + g * 8;
                bf[j] = ld_frag(bp, bp + 16);
            }
        }
#pragma unroll
        for (int mt = 0; mt < 5; ++mt) {
            // A frag: row mt*16+l15, slab cols g*8..+8 and 16+g*8..+8
            const _Float16* ap = &Ah[(mt * 16 + l15) * LDA + g * 8];
            v16h af = ld_frag(ap, ap + 16);
#pragma unroll
            for (int j = 0; j < JMAX; ++j)
                if (w + 8 * j < NTILES)
                    acc[mt][j] = wmma_f16(af, bf[j], acc[mt][j]);
        }
    }

    // epilogue1: write acc transposed into Tt[f][m] as f16
#pragma unroll
    for (int j = 0; j < JMAX; ++j) {
        if (w + 8 * j < NTILES) {
            const int nt = w + 8 * j;
#pragma unroll
            for (int mt = 0; mt < 5; ++mt) {
                v8h hv;
#pragma unroll
                for (int r = 0; r < 8; ++r) hv[r] = (_Float16)acc[mt][j][r];
                // lane holds col f = nt*16+l15, rows m = mt*16+g*8+r contiguous
                *(v8h*)&Tt[(nt * 16 + l15) * LDT + mt * 16 + g * 8] = hv;
            }
        }
    }
    __syncthreads();

    // ---------------- GEMM2: out[n][f] = att @ tmp ----------------
#pragma unroll
    for (int mt = 0; mt < 5; ++mt)
#pragma unroll
        for (int j = 0; j < JMAX; ++j)
#pragma unroll
            for (int r = 0; r < 8; ++r) acc[mt][j][r] = 0.f;

#pragma unroll
    for (int ks = 0; ks < 3; ++ks) {     // K = 96 (padded node dim)
        const int kb = ks * 32;
        v16h bf[JMAX];
#pragma unroll
        for (int j = 0; j < JMAX; ++j) {
            if (w + 8 * j < NTILES) {
                const int nt = w + 8 * j;
                const _Float16* bp = &Tt[(nt * 16 + l15) * LDT + kb + g * 8];
                bf[j] = ld_frag(bp, bp + 16);
            }
        }
#pragma unroll
        for (int mt = 0; mt < 5; ++mt) {
            const _Float16* ap = attP + (mt * 16 + l15) * MPADK + kb + g * 8;
            v16h af = ld_frag(ap, ap + 16);
#pragma unroll
            for (int j = 0; j < JMAX; ++j)
                if (w + 8 * j < NTILES)
                    acc[mt][j] = wmma_f16(af, bf[j], acc[mt][j]);
        }
    }

    // epilogue2: add bias, scatter to global (rows n < 66, cols f < Nstore)
    float* outb = out + (size_t)b * NNODE * ldout;
#pragma unroll
    for (int j = 0; j < JMAX; ++j) {
        if (w + 8 * j < NTILES) {
            const int nt = w + 8 * j;
            const int f = nt * 16 + l15;
            if (f < Nstore) {
                float bv = bias[f];
#pragma unroll
                for (int mt = 0; mt < 5; ++mt) {
#pragma unroll
                    for (int r = 0; r < 8; ++r) {
                        int n = mt * 16 + g * 8 + r;
                        if (n < NNODE) outb[(size_t)n * ldout + f] = acc[mt][j][r] + bv;
                    }
                }
            }
        }
    }
}

// ---------------------------------------------------------------------------
// BatchNorm (training-mode, biased var) + tanh, per channel over B=512.
// res == nullptr : io = tanh(bn(io))          ; out16 = f16 copy
// res != nullptr : res += tanh(bn(io))        ; out16 = f16 copy of new res
// grid = 66 blocks x 256 threads = 16896 channels
// ---------------------------------------------------------------------------
__global__ __launch_bounds__(256)
void bn_tanh_kernel(float* io, const float* __restrict__ gma,
                    const float* __restrict__ bta, float* res,
                    _Float16* __restrict__ out16)
{
    int c = blockIdx.x * 256 + threadIdx.x;
    if (c >= CCH) return;
    float s = 0.f, s2 = 0.f;
    for (int bb = 0; bb < BATCH; ++bb) {
        float v = io[(size_t)bb * CCH + c];
        s += v; s2 += v * v;
    }
    float mu  = s * (1.f / BATCH);
    float var = fmaxf(s2 * (1.f / BATCH) - mu * mu, 0.f);
    float sc  = gma[c] * rsqrtf(var + 1e-5f);
    float sh  = bta[c];
    for (int bb = 0; bb < BATCH; ++bb) {
        size_t idx = (size_t)bb * CCH + c;
        float o = tanhf((io[idx] - mu) * sc + sh);
        if (res) { float nv = res[idx] + o; res[idx] = nv; out16[idx] = (_Float16)nv; }
        else     { io[idx] = o;             out16[idx] = (_Float16)o; }
    }
}

// ---------------------------------------------------------------------------
// DCT: xd[b,n,d] = sum_t dm[d,t] * x[b,t,n]; also emits f16 copy padded to 32
// ---------------------------------------------------------------------------
__global__ __launch_bounds__(256)
void dct_kernel(const float* __restrict__ x, const float* __restrict__ dm,
                float* __restrict__ xd, _Float16* __restrict__ xd16)
{
    __shared__ float xs[TLEN * NNODE];           // 33 KB: x[b] tile
    int b = blockIdx.x, tid = threadIdx.x;
    const float* xb = x + (size_t)b * TLEN * NNODE;
    for (int i = tid; i < TLEN * NNODE; i += 256) xs[i] = xb[i];
    __syncthreads();
    float*    xdb  = xd   + (size_t)b * NNODE * DCTN;
    _Float16* xdhb = xd16 + (size_t)b * NNODE * DPAD;
    for (int o = tid; o < NNODE * DCTN; o += 256) {
        int n = o / DCTN, d = o - n * DCTN;
        float sum = 0.f;
        for (int t = 0; t < TLEN; ++t) sum += dm[d * TLEN + t] * xs[t * NNODE + n];
        xdb[o] = sum;
        xdhb[n * DPAD + d] = (_Float16)sum;
    }
    // zero the k-padding (d = 20..31) of the f16 tile
    for (int i = tid; i < NNODE * (DPAD - DCTN); i += 256) {
        int n = i / (DPAD - DCTN), d = DCTN + i % (DPAD - DCTN);
        xdhb[n * DPAD + d] = (_Float16)0.f;
    }
}

// ---------------------------------------------------------------------------
// iDCT + xd residual + history splice:
// out[b,t,n] = t<25 ? x[b,t,n] : sum_d im[t,d]*(y[b,n,d]+xd[b,n,d])
// ---------------------------------------------------------------------------
__global__ __launch_bounds__(256)
void idct_kernel(const float* __restrict__ x, const float* __restrict__ im,
                 const float* __restrict__ y, const float* __restrict__ xd,
                 float* __restrict__ out)
{
    __shared__ float rs[NNODE * DCTN];
    int b = blockIdx.x, tid = threadIdx.x;
    const float* yb  = y  + (size_t)b * NNODE * DCTN;
    const float* xdb = xd + (size_t)b * NNODE * DCTN;
    for (int i = tid; i < NNODE * DCTN; i += 256) rs[i] = yb[i] + xdb[i];
    __syncthreads();
    const float* xb = x + (size_t)b * TLEN * NNODE;
    float* ob = out + (size_t)b * TLEN * NNODE;
    for (int o = tid; o < TLEN * NNODE; o += 256) {
        int t = o / NNODE, n = o - t * NNODE;
        float v;
        if (t < THIS) v = xb[o];
        else {
            v = 0.f;
            for (int d = 0; d < DCTN; ++d) v += im[t * TLEN + d] * rs[n * DCTN + d];
        }
        ob[o] = v;
    }
}

// ---------------------------------------------------------------------------
// Weight prep: wT[n][k] = (f16) w[k][n], zero-padded to [Npad][Kpad]
// ---------------------------------------------------------------------------
__global__ __launch_bounds__(256)
void prep_wT(const float* __restrict__ w, _Float16* __restrict__ wT,
             int Kin, int N, int Kpad, int Npad)
{
    int i = blockIdx.x * 256 + threadIdx.x;
    if (i >= Npad * Kpad) return;
    int n = i / Kpad, k = i - n * Kpad;
    float v = (n < N && k < Kin) ? w[(size_t)k * N + n] : 0.f;
    wT[i] = (_Float16)v;
}

// att prep: attP[80][96] f16, zero-padded from [66][66]
__global__ __launch_bounds__(256)
void prep_att(const float* __restrict__ att, _Float16* __restrict__ attP)
{
    int i = blockIdx.x * 256 + threadIdx.x;
    if (i >= MPADR * MPADK) return;
    int n = i / MPADK, m = i - n * MPADK;
    attP[i] = (_Float16)((n < NNODE && m < NNODE) ? att[n * NNODE + m] : 0.f);
}

// ---------------------------------------------------------------------------
extern "C" void kernel_launch(void* const* d_in, const int* in_sizes, int n_in,
                              void* d_out, int out_size, void* d_ws, size_t ws_size,
                              hipStream_t stream)
{
    const float* x         = (const float*)d_in[0];
    const float* trans_m   = (const float*)d_in[1];
    const float* itrans_m  = (const float*)d_in[2];
    const float* enc_w     = (const float*)d_in[3];
    const float* enc_att   = (const float*)d_in[4];
    const float* enc_b     = (const float*)d_in[5];
    const float* bn0_g     = (const float*)d_in[6];
    const float* bn0_b     = (const float*)d_in[7];
    const float* blk1_w    = (const float*)d_in[8];
    const float* blk1_att  = (const float*)d_in[9];
    const float* blk1_b    = (const float*)d_in[10];
    const float* blk1_bg   = (const float*)d_in[11];
    const float* blk1_bb   = (const float*)d_in[12];
    const float* blk2_w    = (const float*)d_in[13];
    const float* blk2_att  = (const float*)d_in[14];
    const float* blk2_b    = (const float*)d_in[15];
    const float* blk2_bg   = (const float*)d_in[16];
    const float* blk2_bb   = (const float*)d_in[17];
    const float* dec_w     = (const float*)d_in[18];
    const float* dec_att   = (const float*)d_in[19];
    const float* dec_b     = (const float*)d_in[20];
    float* out = (float*)d_out;

    // workspace layout (bytes); ~111 MB total
    char* ws = (char*)d_ws;
    const size_t XD_BYTES  = (size_t)BATCH * NNODE * DCTN * 4;   //  2,703,360
    const size_t H_BYTES   = (size_t)BATCH * NNODE * HID  * 4;   // 34,603,008
    const size_t XD16_B    = (size_t)BATCH * NNODE * DPAD * 2;   //  2,162,688
    const size_t H16_B     = (size_t)BATCH * NNODE * HID  * 2;   // 17,301,504
    size_t off = 0;
    float*    xd   = (float*)(ws + off);    off += XD_BYTES;
    float*    h    = (float*)(ws + off);    off += H_BYTES;
    float*    t    = (float*)(ws + off);    off += H_BYTES;
    _Float16* xd16 = (_Float16*)(ws + off); off += XD16_B;
    _Float16* h16  = (_Float16*)(ws + off); off += H16_B;
    _Float16* t16  = (_Float16*)(ws + off); off += H16_B;
    _Float16* encT = (_Float16*)(ws + off);
    _Float16* b1T  = encT + 256 * DPAD;
    _Float16* b2T  = b1T  + NSTAGE * HID * HID;
    _Float16* decT = b2T  + NSTAGE * HID * HID;
    _Float16* attE = decT + 32 * HID;
    _Float16* att1 = attE + MPADR * MPADK;
    _Float16* att2 = att1 + NSTAGE * MPADR * MPADK;
    _Float16* attD = att2 + NSTAGE * MPADR * MPADK;

    // ---- weight prep (cheap, deterministic) ----
    prep_wT<<<32, 256, 0, stream>>>(enc_w, encT, DCTN, HID, DPAD, HID);
    for (int s = 0; s < NSTAGE; ++s) {
        prep_wT<<<256, 256, 0, stream>>>(blk1_w + (size_t)s * HID * HID,
                                         b1T + (size_t)s * HID * HID, HID, HID, HID, HID);
        prep_wT<<<256, 256, 0, stream>>>(blk2_w + (size_t)s * HID * HID,
                                         b2T + (size_t)s * HID * HID, HID, HID, HID, HID);
    }
    prep_wT<<<32, 256, 0, stream>>>(dec_w, decT, HID, DCTN, HID, 32);
    prep_att<<<30, 256, 0, stream>>>(enc_att, attE);
    for (int s = 0; s < NSTAGE; ++s) {
        prep_att<<<30, 256, 0, stream>>>(blk1_att + (size_t)s * NNODE * NNODE,
                                         att1 + (size_t)s * MPADR * MPADK);
        prep_att<<<30, 256, 0, stream>>>(blk2_att + (size_t)s * NNODE * NNODE,
                                         att2 + (size_t)s * MPADR * MPADK);
    }
    prep_att<<<30, 256, 0, stream>>>(dec_att, attD);

    // ---- forward pipeline ----
    dct_kernel<<<BATCH, 256, 0, stream>>>(x, trans_m, xd, xd16);

    // encoder GC -> h ; BN+tanh in place (+ f16 feed h16)
    gc_kernel<16><<<BATCH, 256, 0, stream>>>(xd16, DPAD, encT, attE, enc_b,
                                             h, HID, HID);
    bn_tanh_kernel<<<66, 256, 0, stream>>>(h, bn0_g, bn0_b, nullptr, h16);

    for (int s = 0; s < NSTAGE; ++s) {
        gc_kernel<16><<<BATCH, 256, 0, stream>>>(h16, HID,
                                                 b1T + (size_t)s * HID * HID,
                                                 att1 + (size_t)s * MPADR * MPADK,
                                                 blk1_b + (size_t)s * HID,
                                                 t, HID, HID);
        bn_tanh_kernel<<<66, 256, 0, stream>>>(t, blk1_bg + (size_t)s * CCH,
                                               blk1_bb + (size_t)s * CCH,
                                               nullptr, t16);
        gc_kernel<16><<<BATCH, 256, 0, stream>>>(t16, HID,
                                                 b2T + (size_t)s * HID * HID,
                                                 att2 + (size_t)s * MPADR * MPADK,
                                                 blk2_b + (size_t)s * HID,
                                                 t, HID, HID);
        bn_tanh_kernel<<<66, 256, 0, stream>>>(t, blk2_bg + (size_t)s * CCH,
                                               blk2_bb + (size_t)s * CCH,
                                               h, h16);   // h += tanh(bn(t)); h16 = f16(h)
    }

    // decoder GC: h16 -> t viewed as [B,66,20]
    gc_kernel<2><<<BATCH, 256, 0, stream>>>(h16, HID, decT, attD, dec_b,
                                            t, DCTN, DCTN);

    // iDCT(y + xd) + history splice
    idct_kernel<<<BATCH, 256, 0, stream>>>(x, itrans_m, t, xd, out);
}
